// ScaledDotProductAttention_21492016349280
// MI455X (gfx1250) — matrix operand order repaired and checked by reading, not compiled, not run on hardware
//
#include <hip/hip_runtime.h>
#include <hip/hip_bf16.h>

// ---------------------------------------------------------------------------
// Causal scaled-dot-product attention forward for MI455X (gfx1250, wave32).
// Outputs BOTH context [B,H,S,DK] and attn [B,H,S,S] (reference returns tuple).
//
// Bandwidth-bound on the 1.07 GB attn store -> two-pass softmax, attn written
// exactly once. Compute: v_wmma_f32_16x16x32_bf16, fragments fed by
// contiguous ds_load_b128 (LDS rows padded to 272 B for 16-B alignment).
// ---------------------------------------------------------------------------

#define B_   4
#define H_   16
#define S_   2048
#define DK_  128
#define QT_  128            // query rows per block (8 waves x 16 rows)
#define KT_  128            // key rows per LDS tile
#define LS_  136            // LDS row stride (halfwords): 272 B, 16-B aligned,
                            // 68-dword stride -> conflict-minimal b128 reads

typedef __attribute__((ext_vector_type(16))) __bf16        v16bf;
typedef __attribute__((ext_vector_type(8)))  float         v8f;
typedef __attribute__((ext_vector_type(8)))  unsigned int  v8u;

__device__ __forceinline__ unsigned short f2bf(float x) {
    unsigned int u = __float_as_uint(x);
    return (unsigned short)((u + 0x7FFFu + ((u >> 16) & 1u)) >> 16);  // RNE
}
__device__ __forceinline__ unsigned int pack2bf(float lo, float hi) {
    return (unsigned int)f2bf(lo) | ((unsigned int)f2bf(hi) << 16);
}

// B-matrix 32x16 bf16 fragment: per lane, K = {2j + 16*half} pairs = 16
// contiguous bf16 (32 B) starting at halfword offset 16*half within the row.
__device__ __forceinline__ v16bf ld_frag_b(const unsigned short* p) {
    union { v8u v; uint4 q[2]; } t;
    t.q[0] = *(const uint4*)(p);
    t.q[1] = *(const uint4*)(p + 8);
    return __builtin_bit_cast(v16bf, t.v);
}

// A-matrix 16x32 bf16 fragment: per lane two contiguous 16-B chunks,
// K = 16h..16h+7 and K = 16h+8..16h+15 (halfword offsets 16h and 16h+8).
__device__ __forceinline__ v16bf ld_frag_a(const unsigned short* rowp, int h) {
    union { v8u v; uint4 q[2]; } t;
    t.q[0] = *(const uint4*)(rowp + 16 * h);
    t.q[1] = *(const uint4*)(rowp + 16 * h + 8);
    return __builtin_bit_cast(v16bf, t.v);
}

__device__ __forceinline__ float rowmax16(float v) {
    #pragma unroll
    for (int off = 1; off < 16; off <<= 1) v = fmaxf(v, __shfl_xor(v, off, 32));
    return v;
}
__device__ __forceinline__ float rowsum16(float v) {
    #pragma unroll
    for (int off = 1; off < 16; off <<= 1) v += __shfl_xor(v, off, 32);
    return v;
}

__device__ __forceinline__ v8f wmma_bf16(v16bf a, v16bf b, v8f c) {
    return __builtin_amdgcn_wmma_f32_16x16x32_bf16(
        false, a, false, b, (short)0, c, false, false);
}

extern __shared__ unsigned short smem[];

__global__ __launch_bounds__(256)
void sdpa_causal_fwd_kernel(const float* __restrict__ Q,
                            const float* __restrict__ K,
                            const float* __restrict__ V,
                            float* __restrict__ ctx,
                            float* __restrict__ attn) {
    const int qtile  = blockIdx.x;           // 0..15 (S/128)
    const int bh     = blockIdx.y;           // 0..63 (B*H)
    const int tid    = threadIdx.x;
    const int lane   = tid & 31;
    const int w      = tid >> 5;             // wave 0..7
    const int ln     = lane & 15;            // N / M-row selector
    const int h      = (lane >> 4) & 1;      // lane half

    unsigned short* Klds = smem;                         // [128][136] (key, dk)
    unsigned short* VT   = smem + KT_ * LS_;             // [128][136] (dk, key)
    unsigned short* Pw   = smem + 2 * KT_ * LS_ + w * 16 * LS_; // per-wave 16x136

    const int qbase0 = qtile * QT_;
    const int qb     = qbase0 + w * 16;      // this wave's first query row

    const size_t head_qkv = (size_t)bh * S_ * DK_;
    const float* Qg = Q + head_qkv + (size_t)(qb + ln) * DK_;
    const float  SC = 0.08838834764831845f;  // 1/sqrt(128)

    // ---- Q A-fragments: K span of frag f is dk = 32f..32f+31.
    // Per lane: 8 floats at 32f+16h and 8 floats at 32f+16h+8 (two float4 pairs).
    v16bf aq[4];
    #pragma unroll
    for (int f = 0; f < 4; ++f) {
        const float* p0 = Qg + 32 * f + 16 * h;
        const float* p1 = Qg + 32 * f + 16 * h + 8;
        float4 a0 = *(const float4*)(p0);
        float4 a1 = *(const float4*)(p0 + 4);
        float4 b0 = *(const float4*)(p1);
        float4 b1 = *(const float4*)(p1 + 4);
        v8u u;
        u[0] = pack2bf(a0.x * SC, a0.y * SC); u[1] = pack2bf(a0.z * SC, a0.w * SC);
        u[2] = pack2bf(a1.x * SC, a1.y * SC); u[3] = pack2bf(a1.z * SC, a1.w * SC);
        u[4] = pack2bf(b0.x * SC, b0.y * SC); u[5] = pack2bf(b0.z * SC, b0.w * SC);
        u[6] = pack2bf(b1.x * SC, b1.y * SC); u[7] = pack2bf(b1.z * SC, b1.w * SC);
        aq[f] = __builtin_bit_cast(v16bf, u);
    }

    float m[8], l[8];
    #pragma unroll
    for (int r = 0; r < 8; ++r) { m[r] = -1e30f; l[r] = 0.0f; }

    // ======================= PASS 1: row max & sumexp =======================
    for (int kt = 0; kt <= qtile; ++kt) {
        __syncthreads();
        const float* Kg = K + head_qkv + (size_t)kt * KT_ * DK_;
        #pragma unroll
        for (int it = 0; it < 16; ++it) {             // 4096 float4 / 256 thr
            int f4  = tid + it * 256;
            int row = f4 >> 5;
            int dk  = (f4 & 31) * 4;
            float4 v = *(const float4*)(Kg + row * DK_ + dk);
            *(unsigned int*)&Klds[row * LS_ + dk]     = pack2bf(v.x, v.y);
            *(unsigned int*)&Klds[row * LS_ + dk + 2] = pack2bf(v.z, v.w);
        }
        __syncthreads();
        if (kt < qtile)
            __builtin_prefetch(Kg + (size_t)KT_ * DK_ + (tid << 4), 0, 0);

        v8f sc[8];
        #pragma unroll
        for (int t = 0; t < 8; ++t) {                 // 8 N-tiles of 16 keys
            const unsigned short* krow = Klds + (t * 16 + ln) * LS_ + 16 * h;
            v16bf bf0 = ld_frag_b(krow);
            v16bf bf1 = ld_frag_b(krow + 32);
            v16bf bf2 = ld_frag_b(krow + 64);
            v16bf bf3 = ld_frag_b(krow + 96);
            v8f acc = {};
            acc = wmma_bf16(aq[0], bf0, acc);
            acc = wmma_bf16(aq[1], bf1, acc);
            acc = wmma_bf16(aq[2], bf2, acc);
            acc = wmma_bf16(aq[3], bf3, acc);
            sc[t] = acc;
        }
        if (kt == qtile) {                            // diagonal tile: mask
            #pragma unroll
            for (int t = 0; t < 8; ++t)
                #pragma unroll
                for (int r = 0; r < 8; ++r) {
                    int qrow = qb + r + 8 * h;
                    int kcol = kt * KT_ + t * 16 + ln;
                    if (kcol > qrow) sc[t][r] = -1e9f;
                }
        }
        #pragma unroll
        for (int r = 0; r < 8; ++r) {                 // one update per k-tile
            float tm = sc[0][r];
            #pragma unroll
            for (int t = 1; t < 8; ++t) tm = fmaxf(tm, sc[t][r]);
            tm = rowmax16(tm);
            float mn = fmaxf(m[r], tm);
            float se = 0.0f;
            #pragma unroll
            for (int t = 0; t < 8; ++t) se += __expf(sc[t][r] - mn);
            se = rowsum16(se);
            l[r] = l[r] * __expf(m[r] - mn) + se;
            m[r] = mn;
        }
    }

    float invl[8];
    #pragma unroll
    for (int r = 0; r < 8; ++r) invl[r] = __builtin_amdgcn_rcpf(l[r]);

    // ============== PASS 2: attn = exp(s-m)/l, ctx += attn @ V ==============
    v8f cacc[8];
    #pragma unroll
    for (int d = 0; d < 8; ++d) cacc[d] = (v8f){};

    for (int kt = 0; kt <= qtile; ++kt) {
        __syncthreads();
        const float* Kg = K + head_qkv + (size_t)kt * KT_ * DK_;
        const float* Vg = V + head_qkv + (size_t)kt * KT_ * DK_;
        #pragma unroll
        for (int it = 0; it < 16; ++it) {
            int f4  = tid + it * 256;
            int row = f4 >> 5;
            int dk  = (f4 & 31) * 4;
            float4 kv = *(const float4*)(Kg + row * DK_ + dk);
            *(unsigned int*)&Klds[row * LS_ + dk]     = pack2bf(kv.x, kv.y);
            *(unsigned int*)&Klds[row * LS_ + dk + 2] = pack2bf(kv.z, kv.w);
            float4 vv = *(const float4*)(Vg + row * DK_ + dk);
            VT[(dk + 0) * LS_ + row] = f2bf(vv.x);    // transpose into LDS
            VT[(dk + 1) * LS_ + row] = f2bf(vv.y);
            VT[(dk + 2) * LS_ + row] = f2bf(vv.z);
            VT[(dk + 3) * LS_ + row] = f2bf(vv.w);
        }
        __syncthreads();
        if (kt < qtile)
            __builtin_prefetch(Kg + (size_t)KT_ * DK_ + (tid << 4), 0, 0);

        // --- scores -> probabilities; write attn + stage P in LDS
        for (int t = 0; t < 8; ++t) {
            const unsigned short* krow = Klds + (t * 16 + ln) * LS_ + 16 * h;
            v16bf bf0 = ld_frag_b(krow);
            v16bf bf1 = ld_frag_b(krow + 32);
            v16bf bf2 = ld_frag_b(krow + 64);
            v16bf bf3 = ld_frag_b(krow + 96);
            v8f acc = {};
            acc = wmma_bf16(aq[0], bf0, acc);
            acc = wmma_bf16(aq[1], bf1, acc);
            acc = wmma_bf16(aq[2], bf2, acc);
            acc = wmma_bf16(aq[3], bf3, acc);
            if (kt == qtile) {
                #pragma unroll
                for (int r = 0; r < 8; ++r) {
                    int qrow = qb + r + 8 * h;
                    int kcol = kt * KT_ + t * 16 + ln;
                    if (kcol > qrow) acc[r] = -1e9f;
                }
            }
            #pragma unroll
            for (int r = 0; r < 8; ++r) {
                float p = __expf(acc[r] - m[r]) * invl[r];   // masked -> exactly 0
                int row_local = r + 8 * h;                   // C: M = r + 8*half
                int col       = t * 16 + ln;                 // N = lane%16
                size_t qrow   = (size_t)(qb + row_local);
                attn[((size_t)bh * S_ + qrow) * S_ + (size_t)kt * KT_ + col] = p;
                Pw[row_local * LS_ + col] = f2bf(p);
            }
        }

        // --- context += P (16x128) @ V (128x128)
        const unsigned short* prow = Pw + ln * LS_;
        v16bf pa[4];                                  // invariant across d-tiles
        #pragma unroll
        for (int f = 0; f < 4; ++f) pa[f] = ld_frag_a(prow + 32 * f, h);

        for (int d = 0; d < 8; ++d) {
            const unsigned short* vrow = VT + (d * 16 + ln) * LS_ + 16 * h;
            v16bf vf0 = ld_frag_b(vrow);
            v16bf vf1 = ld_frag_b(vrow + 32);
            v16bf vf2 = ld_frag_b(vrow + 64);
            v16bf vf3 = ld_frag_b(vrow + 96);
            v8f c = cacc[d];
            c = wmma_bf16(pa[0], vf0, c);
            c = wmma_bf16(pa[1], vf1, c);
            c = wmma_bf16(pa[2], vf2, c);
            c = wmma_bf16(pa[3], vf3, c);
            cacc[d] = c;
        }
    }

    // ---- store context
    float* cg = ctx + head_qkv + (size_t)qb * DK_;
    #pragma unroll
    for (int d = 0; d < 8; ++d) {
        #pragma unroll
        for (int r = 0; r < 8; ++r) {
            cg[(r + 8 * h) * DK_ + d * 16 + ln] = cacc[d][r];
        }
    }

    // ---- zero-fill fully-masked attn region: cols >= (qtile+1)*128
    const int cstart = (qtile + 1) * KT_;
    const int ncols  = S_ - cstart;                   // multiple of 128
    if (ncols > 0) {
        float* arow = attn + ((size_t)bh * S_ + qbase0) * S_ + cstart;
        const float4 z = make_float4(0.f, 0.f, 0.f, 0.f);
        for (int row = w; row < QT_; row += 8) {      // one wave per row group
            float* p = arow + (size_t)row * S_;
            for (int c = lane * 4; c < ncols; c += 128)
                *(float4*)(p + c) = z;
        }
    }
}

extern "C" void kernel_launch(void* const* d_in, const int* in_sizes, int n_in,
                              void* d_out, int out_size, void* d_ws, size_t ws_size,
                              hipStream_t stream) {
    (void)in_sizes; (void)n_in; (void)out_size; (void)d_ws; (void)ws_size;
    const float* Q = (const float*)d_in[0];
    const float* K = (const float*)d_in[1];
    const float* V = (const float*)d_in[2];
    // d_in[3] = attn_mask (causal) — implemented analytically in-kernel.

    float* ctx  = (float*)d_out;
    float* attn = (float*)d_out + (size_t)B_ * H_ * S_ * DK_;

    // LDS: K tile + transposed V tile + 8 per-wave P scratch tiles (bf16).
    const size_t smem_bytes =
        (size_t)(2 * KT_ * LS_ + 8 * 16 * LS_) * sizeof(unsigned short);

    dim3 grid(S_ / QT_, B_ * H_);
    dim3 block(256);
    sdpa_causal_fwd_kernel<<<grid, block, smem_bytes, stream>>>(Q, K, V, ctx, attn);
}
